// DPMambaBlock_75720273429275
// MI455X (gfx1250) — matrix operand order repaired and checked by reading, not compiled, not run on hardware
//
#include <hip/hip_runtime.h>

// ---------------------------------------------------------------------------
// MI455X (gfx1250) implementation of the DPMamba block.
// GEMMs use v_wmma_f32_16x16x32_f16 (wave32); LDS tiles are laid out so every
// WMMA fragment load is a contiguous 16B ds_load_b128 (A row-major, B stored
// transposed).  Activations are packed f16 channels-last; the selective scan
// uses 16-lane-per-channel wave32 shfl reductions.
// ---------------------------------------------------------------------------

typedef _Float16 f16;
typedef _Float16 v16h __attribute__((ext_vector_type(16)));
typedef _Float16 v8h  __attribute__((ext_vector_type(8)));
typedef _Float16 v2h  __attribute__((ext_vector_type(2)));
typedef float    v8f  __attribute__((ext_vector_type(8)));

constexpr int Bn  = 8;
constexpr int Cn  = 96;
constexpr int Hn  = 64;
constexpr int Wn  = 64;
constexpr int HWn = Hn * Wn;          // 4096
constexpr int M1n = Bn * HWn;         // 32768 rows (b, p)
constexpr int DIn = 192;              // mamba inner dim
constexpr float BNs = 1e-5f;

__device__ __forceinline__ float siluf(float x){ return x / (1.f + __expf(-x)); }
__device__ __forceinline__ float sigf(float x){ return 1.f / (1.f + __expf(-x)); }

// ------------------------- generic packed-f16 GEMM -------------------------
// C[M,N] = A[M,K] * B[K,N] (+bias[n]); A,B f16 row-major zero-padded to
// (ceil64(M) x lda) and (Kp x ldb with Np=ceil64(N) cols readable). K mult 32.
__global__ __launch_bounds__(128)
void k_gemm(const f16* __restrict__ A, const f16* __restrict__ Bw,
            const float* __restrict__ bias, float* __restrict__ Co,
            int M, int N, int K, int lda, int ldb, int ldc,
            long long sA, long long sB, long long sC)
{
  A  += (long long)blockIdx.z * sA;
  Bw += (long long)blockIdx.z * sB;
  Co += (long long)blockIdx.z * sC;

  // Row stride 40 halves = 80B keeps every 8-half fragment run 16B aligned.
  __shared__ f16 As[64][40];   // A tile, row-major  (m, k)
  __shared__ f16 Bt[64][40];   // B tile, TRANSPOSED (n, k)

  const int tid  = threadIdx.x;
  const int wave = tid >> 5;
  const int lane = tid & 31;
  const int m0 = blockIdx.x * 64;
  const int n0 = blockIdx.y * 64;

  v8f acc[4];
#pragma unroll
  for (int j = 0; j < 4; ++j)
#pragma unroll
    for (int r = 0; r < 8; ++r) acc[j][r] = 0.f;

  union HU { v16h v; v8h h[2]; };

  const int ksteps = K >> 5;
  for (int kt = 0; kt < ksteps; ++kt) {
    const int k0 = kt << 5;

    // stage A tile: 64 rows x 32 halves as 16B vectors (2 per thread)
    for (int i = tid; i < 64 * 4; i += 128) {
      int r = i >> 2, cb = (i & 3) * 8;
      *(v8h*)&As[r][cb] = *(const v8h*)(A + (long long)(m0 + r) * lda + k0 + cb);
    }
    // stage B tile transposed: read 8 n-consecutive halves, scatter into Bt
    for (int i = tid; i < 32 * 8; i += 128) {
      int k = i >> 3, nb8 = (i & 7) * 8;
      v8h tmp = *(const v8h*)(Bw + (long long)(k0 + k) * ldb + n0 + nb8);
#pragma unroll
      for (int j = 0; j < 8; ++j) Bt[nb8 + j][k] = tmp[j];
    }
    // prefetch next K-step tiles into cache while we compute this one
    if (kt + 1 < ksteps) {
      __builtin_prefetch(A + (long long)(m0 + (tid & 63)) * lda + k0 + 32, 0, 1);
      __builtin_prefetch(Bw + (long long)(k0 + 32 + (tid & 31)) * ldb + n0, 0, 1);
    }
    __syncthreads();

    // A fragment (16x32): lane m=lane%16, two contiguous 8-half runs
    const int arow = wave * 16 + (lane & 15);
    const int kb   = (lane >> 4) * 8;
    HU af;
    af.h[0] = *(const v8h*)&As[arow][kb];        // K = kb..kb+7
    af.h[1] = *(const v8h*)&As[arow][16 + kb];   // K = 16+kb..16+kb+7

    const int kb2 = (lane >> 4) * 8;
#pragma unroll
    for (int j = 0; j < 4; ++j) {
      const int nrow = j * 16 + (lane & 15);
      HU bf;
      bf.h[0] = *(const v8h*)&Bt[nrow][kb2];      // K = kb2..kb2+7
      bf.h[1] = *(const v8h*)&Bt[nrow][16 + kb2]; // K = 16+kb2..16+kb2+7
      acc[j] = __builtin_amdgcn_wmma_f32_16x16x32_f16(
          false, af.v, false, bf.v, (short)0, acc[j], false, false);
    }
    __syncthreads();
  }

  const int wrow = wave * 16;
#pragma unroll
  for (int j = 0; j < 4; ++j) {
    const int nn = n0 + j * 16 + (lane & 15);
#pragma unroll
    for (int r = 0; r < 8; ++r) {
      int mm = m0 + wrow + r + ((lane >> 4) << 3);
      if (mm < M && nn < N) {
        float v = acc[j][r];
        if (bias) v += bias[nn];
        Co[(long long)mm * ldc + nn] = v;
      }
    }
  }
}

// ------------------------- pack fp32 -> padded f16 -------------------------
__global__ void k_pack(const float* __restrict__ src, f16* __restrict__ dst,
                       int R, int Cc, long long sr, long long sc,
                       int Rp, int Cp, long long sb, long long db)
{
  src += (long long)blockIdx.z * sb;
  dst += (long long)blockIdx.z * db;
  long long i = (long long)blockIdx.x * blockDim.x + threadIdx.x;
  long long tot = (long long)Rp * Cp;
  if (i >= tot) return;
  int r = (int)(i / Cp), c = (int)(i % Cp);
  float v = (r < R && c < Cc) ? src[(long long)r * sr + (long long)c * sc] : 0.f;
  dst[i] = (f16)v;
}

// ------------------------- depthwise 3x3 convs -----------------------------
__global__ void k_dw3x3_nchw(const float* __restrict__ x, const float* __restrict__ w,
                             const float* __restrict__ bias, f16* __restrict__ out)
{
  long long i = (long long)blockIdx.x * blockDim.x + threadIdx.x;
  if (i >= (long long)M1n * Cn) return;
  int c = (int)(i % Cn);
  long long m = i / Cn;
  int b = (int)(m / HWn), p = (int)(m % HWn);
  int y = p >> 6, xx = p & 63;
  const float* plane = x + (long long)(b * Cn + c) * HWn;
  float s = bias[c];
  for (int ky = 0; ky < 3; ++ky) {
    int yy = y + ky - 1; if (yy < 0 || yy >= Hn) continue;
    for (int kx = 0; kx < 3; ++kx) {
      int xk = xx + kx - 1; if (xk < 0 || xk >= Wn) continue;
      s += plane[yy * Wn + xk] * w[c * 9 + ky * 3 + kx];
    }
  }
  out[i] = (f16)s;
}

__global__ void k_dw3x3_rows(const float* __restrict__ in, const float* __restrict__ w,
                             const float* __restrict__ bias, f16* __restrict__ out)
{
  long long i = (long long)blockIdx.x * blockDim.x + threadIdx.x;
  if (i >= (long long)M1n * Cn) return;
  int c = (int)(i % Cn);
  long long m = i / Cn;
  int b = (int)(m / HWn), p = (int)(m % HWn);
  int y = p >> 6, xx = p & 63;
  float s = bias[c];
  for (int ky = 0; ky < 3; ++ky) {
    int yy = y + ky - 1; if (yy < 0 || yy >= Hn) continue;
    for (int kx = 0; kx < 3; ++kx) {
      int xk = xx + kx - 1; if (xk < 0 || xk >= Wn) continue;
      s += in[((long long)(b * HWn + yy * Wn + xk)) * Cn + c] * w[c * 9 + ky * 3 + kx];
    }
  }
  out[i] = (f16)s;
}

// ------------------------- simple elementwise ------------------------------
__global__ void k_bn_rows(float* __restrict__ a, const float* __restrict__ g,
                          const float* __restrict__ bt, long long n, int Cch)
{
  long long i = (long long)blockIdx.x * blockDim.x + threadIdx.x;
  if (i >= n) return;
  int c = (int)(i % Cch);
  a[i] = a[i] * (g[c] * rsqrtf(1.f + BNs)) + bt[c];
}

__global__ void k_softplus(float* __restrict__ d, long long n)
{
  long long i = (long long)blockIdx.x * blockDim.x + threadIdx.x;
  if (i >= n) return;
  float x = d[i];
  d[i] = (x > 20.f) ? x : log1pf(__expf(x));
}

// ------------------------- layernorm (wave32 per row) ----------------------
__global__ __launch_bounds__(128)
void k_layernorm(const float* __restrict__ src, f16* __restrict__ dst,
                 const float* __restrict__ g, const float* __restrict__ bt,
                 int rows, int rpb, int transposed)
{
  int wave = threadIdx.x >> 5;
  int lane = threadIdx.x & 31;
  int row = blockIdx.x * 4 + wave;
  if (row >= rows) return;
  long long srow = row;
  if (transposed) {
    int b = row / rpb, l = row % rpb;
    int p = (l & 63) * 64 + (l >> 6);       // l = x*H + y  ->  p = y*W + x
    srow = (long long)b * rpb + p;
  }
  const float* s = src + srow * Cn;
  float v0 = s[lane], v1 = s[lane + 32], v2 = s[lane + 64];
  float sum = v0 + v1 + v2;
  float sq  = v0 * v0 + v1 * v1 + v2 * v2;
#pragma unroll
  for (int m = 16; m >= 1; m >>= 1) {
    sum += __shfl_xor(sum, m, 32);
    sq  += __shfl_xor(sq,  m, 32);
  }
  float mean = sum * (1.f / 96.f);
  float var  = sq * (1.f / 96.f) - mean * mean;
  float rstd = rsqrtf(var + 1e-5f);
  f16* d = dst + (long long)row * Cn;
  d[lane]      = (f16)((v0 - mean) * rstd * g[lane]      + bt[lane]);
  d[lane + 32] = (f16)((v1 - mean) * rstd * g[lane + 32] + bt[lane + 32]);
  d[lane + 64] = (f16)((v2 - mean) * rstd * g[lane + 64] + bt[lane + 64]);
}

// ------------------------- mamba pieces ------------------------------------
__global__ void k_conv1d_silu(const float* __restrict__ xz, const float* __restrict__ w,
                              const float* __restrict__ bias, float* __restrict__ u,
                              f16* __restrict__ u16, int L)
{
  long long i = (long long)blockIdx.x * blockDim.x + threadIdx.x;
  if (i >= (long long)Bn * L * DIn) return;
  int c = (int)(i % DIn);
  long long t = i / DIn;
  int b = (int)(t / L), l = (int)(t % L);
  float s = bias[c];
  for (int kk = 0; kk < 4; ++kk) {
    int ls = l - 3 + kk; if (ls < 0) continue;
    s += xz[((long long)(b * L + ls)) * 384 + c] * w[c * 4 + kk];
  }
  float v = siluf(s);
  u[i] = v;
  u16[i] = (f16)v;
}

__global__ void k_dtprep(const float* __restrict__ dbc, f16* __restrict__ dtin,
                         float* __restrict__ Bm, float* __restrict__ Cm, int rows)
{
  long long i = (long long)blockIdx.x * blockDim.x + threadIdx.x;
  if (i >= (long long)rows * 64) return;
  long long m = i >> 6;
  int j = (int)(i & 63);
  const float* r = dbc + m * 38;
  if (j < 32)      dtin[m * 32 + j]        = (f16)((j < 6) ? r[j] : 0.f);
  else if (j < 48) Bm[m * 16 + (j - 32)]   = r[6 + (j - 32)];
  else             Cm[m * 16 + (j - 48)]   = r[22 + (j - 48)];
}

// 16 lanes per (b, channel): lane s holds state s; reduce y via shfl_xor.
__global__ __launch_bounds__(128)
void k_scan(const float* __restrict__ dt, const float* __restrict__ u,
            const float* __restrict__ Bm, const float* __restrict__ Cm,
            const float* __restrict__ z, const float* __restrict__ Alog,
            const float* __restrict__ Dp, f16* __restrict__ yout, int L)
{
  int g = blockIdx.x * 8 + (threadIdx.x >> 4);
  int s = threadIdx.x & 15;
  int b = g / DIn, c = g % DIn;
  float Av = -__expf(Alog[c * 16 + s]);
  float Dv = Dp[c];
  float h = 0.f;
  const long long rb = (long long)b * L;
  for (int l = 0; l < L; ++l) {
    long long r = rb + l;
    float dtv = dt[r * DIn + c];
    float uv  = u [r * DIn + c];
    float bs  = Bm[r * 16 + s];
    float cs  = Cm[r * 16 + s];
    h = __expf(dtv * Av) * h + (dtv * uv) * bs;
    float part = h * cs;
    part += __shfl_xor(part, 1, 32);
    part += __shfl_xor(part, 2, 32);
    part += __shfl_xor(part, 4, 32);
    part += __shfl_xor(part, 8, 32);
    if (s == 0) {
      float zv = z[r * 384 + c];
      float y = part + uv * Dv;
      yout[r * DIn + c] = (f16)(y * siluf(zv));
    }
  }
}

// ------------------------- scanner (attention) -----------------------------
__global__ __launch_bounds__(256)
void k_chanmean(const float* __restrict__ h, float* __restrict__ avg)
{
  __shared__ float red[256];
  int bc = blockIdx.x; int b = bc / Cn, c = bc % Cn;
  float s = 0.f;
  for (int p = threadIdx.x; p < HWn; p += 256)
    s += h[((long long)(b * HWn + p)) * Cn + c];
  red[threadIdx.x] = s; __syncthreads();
  for (int st = 128; st > 0; st >>= 1) {
    if (threadIdx.x < st) red[threadIdx.x] += red[threadIdx.x + st];
    __syncthreads();
  }
  if (!threadIdx.x) avg[bc] = red[0] * (1.f / HWn);
}

__global__ void k_ca1(const float* avg, const float* w1, const float* b1,
                      const float* pr, float* a1)
{
  int i = blockIdx.x * blockDim.x + threadIdx.x;
  if (i >= Bn * 12) return;
  int b = i / 12, j = i % 12;
  float s = b1[j];
  for (int c = 0; c < Cn; ++c) s += avg[b * Cn + c] * w1[j * Cn + c];
  a1[i] = (s > 0.f) ? s : pr[0] * s;
}

__global__ void k_ca2(const float* a1, const float* w2, const float* b2, float* cf)
{
  int i = blockIdx.x * blockDim.x + threadIdx.x;
  if (i >= Bn * Cn) return;
  int b = i / Cn, c = i % Cn;
  float s = b2[c];
  for (int j = 0; j < 12; ++j) s += a1[b * 12 + j] * w2[c * 12 + j];
  cf[i] = 1.f + sigf(s);
}

__global__ void k_spatial_sum(const float* __restrict__ h, const float* __restrict__ cf,
                              float* __restrict__ s0)
{
  int i = blockIdx.x * blockDim.x + threadIdx.x;
  if (i >= Bn * HWn) return;
  int b = i / HWn;
  const float* row = h + (long long)i * Cn;
  const float* cfb = cf + b * Cn;
  float s = 0.f;
  for (int c = 0; c < Cn; ++c) s += row[c] * cfb[c];
  s0[i] = s;
}

__global__ void k_conv7(const float* __restrict__ s0, const float* __restrict__ w7,
                        const float* __restrict__ g0, const float* __restrict__ b0,
                        const float* __restrict__ pr, float* __restrict__ s1)
{
  int i = blockIdx.x * blockDim.x + threadIdx.x;
  if (i >= Bn * HWn) return;
  int b = i / HWn, p = i % HWn;
  int y = p >> 6, x = p & 63;
  float s = 0.f;
  for (int ky = 0; ky < 7; ++ky) {
    int yy = y + ky - 3; if (yy < 0 || yy >= Hn) continue;
    for (int kx = 0; kx < 7; ++kx) {
      int xk = x + kx - 3; if (xk < 0 || xk >= Wn) continue;
      s += w7[ky * 7 + kx] * s0[b * HWn + yy * Wn + xk];
    }
  }
  float v = s * (g0[0] * rsqrtf(1.f + BNs)) + b0[0];
  s1[i] = (v > 0.f) ? v : pr[0] * v;
}

__global__ void k_conv3_1to4(const float* __restrict__ s1, const float* __restrict__ w,
                             float* __restrict__ out)
{
  int i = blockIdx.x * blockDim.x + threadIdx.x;
  if (i >= Bn * 4 * HWn) return;
  int b = i / (4 * HWn);
  int rem = i % (4 * HWn);
  int o = rem / HWn, p = rem % HWn;
  int y = p >> 6, x = p & 63;
  float s = 0.f;
  for (int ky = 0; ky < 3; ++ky) {
    int yy = y + ky - 1; if (yy < 0 || yy >= Hn) continue;
    for (int kx = 0; kx < 3; ++kx) {
      int xk = x + kx - 1; if (xk < 0 || xk >= Wn) continue;
      s += w[o * 9 + ky * 3 + kx] * s1[b * HWn + yy * Wn + xk];
    }
  }
  out[i] = s;
}

__global__ void k_conv3_4to4(const float* __restrict__ in, const float* __restrict__ w,
                             float* __restrict__ out, int dil)
{
  int i = blockIdx.x * blockDim.x + threadIdx.x;
  if (i >= Bn * 4 * HWn) return;
  int b = i / (4 * HWn);
  int rem = i % (4 * HWn);
  int o = rem / HWn, p = rem % HWn;
  int y = p >> 6, x = p & 63;
  float s = 0.f;
  for (int ic = 0; ic < 4; ++ic) {
    const float* pl = in + (long long)(b * 4 + ic) * HWn;
    for (int ky = 0; ky < 3; ++ky) {
      int yy = y + dil * (ky - 1); if (yy < 0 || yy >= Hn) continue;
      for (int kx = 0; kx < 3; ++kx) {
        int xk = x + dil * (kx - 1); if (xk < 0 || xk >= Wn) continue;
        s += w[((o * 4 + ic) * 3 + ky) * 3 + kx] * pl[yy * Wn + xk];
      }
    }
  }
  out[i] = s;
}

__global__ void k_conv1sig(const float* __restrict__ s4, const float* __restrict__ w1,
                           const float* __restrict__ b1, float* __restrict__ pf)
{
  int i = blockIdx.x * blockDim.x + threadIdx.x;
  if (i >= Bn * HWn) return;
  int b = i / HWn, p = i % HWn;
  float s = b1[0];
  for (int o = 0; o < 4; ++o) s += w1[o] * s4[(long long)(b * 4 + o) * HWn + p];
  pf[i] = 1.f + sigf(s);
}

__global__ void k_pack_xatt(const float* __restrict__ h, const float* __restrict__ cf,
                            const float* __restrict__ pf, f16* __restrict__ xatt)
{
  long long i = (long long)blockIdx.x * blockDim.x + threadIdx.x;
  if (i >= (long long)M1n * Cn) return;
  int c = (int)(i % Cn);
  long long m = i / Cn;
  int b = (int)(m / HWn);
  xatt[i] = (f16)(h[i] * cf[b * Cn + c] * pf[m]);
}

__global__ void k_softmax64(const float* __restrict__ logits, f16* __restrict__ Aout)
{
  int m = blockIdx.x * blockDim.x + threadIdx.x;
  if (m >= M1n) return;
  int b = m / HWn, p = m % HWn;
  const float* lr = logits + (long long)m * 64;
  float mx = lr[0];
  for (int t = 1; t < 64; ++t) mx = fmaxf(mx, lr[t]);
  float sum = 0.f;
  for (int t = 0; t < 64; ++t) sum += __expf(lr[t] - mx);
  float inv = 1.f / sum;
  f16* ab = Aout + (long long)b * 64 * HWn;
  for (int t = 0; t < 64; ++t) ab[t * HWn + p] = (f16)(__expf(lr[t] - mx) * inv);
}

// ------------------------- fusion tail -------------------------------------
__global__ void k_build_cat(const float* __restrict__ x0, const float* __restrict__ x01s,
                            const float* __restrict__ x1b, const float* __restrict__ g1,
                            const float* __restrict__ b1, f16* __restrict__ cat)
{
  long long i = (long long)blockIdx.x * blockDim.x + threadIdx.x;
  if (i >= (long long)M1n * 288) return;
  int j = (int)(i % 288);
  long long m = i / 288;
  int b = (int)(m / HWn), p = (int)(m % HWn);
  float v;
  if (j < 96) {
    v = x0[m * 96 + j];
  } else if (j < 192) {
    int c = j - 96;
    int l = (p & 63) * 64 + (p >> 6);       // seq index for transposed scan
    v = x01s[((long long)(b * HWn + l)) * 96 + c];
  } else {
    int c = j - 192;
    v = x1b[((long long)(b * Cn + c)) * HWn + p];
  }
  cat[i] = (f16)(v * (g1[j] * rsqrtf(1.f + BNs)) + b1[j]);
}

__global__ void k_hm(const float* __restrict__ resm, const float* __restrict__ catc,
                     const float* __restrict__ gM, const float* __restrict__ bM,
                     const float* __restrict__ g2, const float* __restrict__ b2,
                     float* __restrict__ hm)
{
  long long i = (long long)blockIdx.x * blockDim.x + threadIdx.x;
  if (i >= (long long)M1n * Cn) return;
  int c = (int)(i % Cn);
  float rs = rsqrtf(1.f + BNs);
  hm[i] = resm[i] * (gM[c] * rs) + bM[c] + catc[i] * (g2[c] * rs) + b2[c];
}

__global__ void k_final(const float* __restrict__ pw2o, const float* __restrict__ g,
                        const float* __restrict__ bt, const float* __restrict__ x,
                        float* __restrict__ out)
{
  long long i = (long long)blockIdx.x * blockDim.x + threadIdx.x;
  if (i >= (long long)Bn * Cn * HWn) return;
  int b = (int)(i / (Cn * HWn));
  int rem = (int)(i % (Cn * HWn));
  int c = rem / HWn, p = rem % HWn;
  float v = pw2o[((long long)(b * HWn + p)) * Cn + c];
  out[i] = v * (g[c] * rsqrtf(1.f + BNs)) + bt[c] + x[i];
}

// ===========================================================================
extern "C" void kernel_launch(void* const* d_in, const int* in_sizes, int n_in,
                              void* d_out, int out_size, void* d_ws, size_t ws_size,
                              hipStream_t stream)
{
  (void)in_sizes; (void)n_in; (void)out_size; (void)ws_size;
  const float* X = (const float*)d_in[0];
  auto P = [&](int i) { return (const float*)d_in[i]; };

  char* base = (char*)d_ws;
  size_t off = 0;
  auto alloc = [&](size_t bytes) -> void* {
    void* p = base + off;
    off += (bytes + 255) & ~(size_t)255;
    return p;
  };

  f16*   dw1h   = (f16*)  alloc((size_t)M1n * 96 * 2);
  float* hrows  = (float*)alloc((size_t)M1n * 96 * 4);
  f16*   hf16   = (f16*)  alloc((size_t)M1n * 96 * 2);
  float* x0r    = (float*)alloc((size_t)M1n * 96 * 4);
  float* x01s   = (float*)alloc((size_t)M1n * 96 * 4);
  float* tokens = (float*)alloc((size_t)512 * 96 * 4);
  float* tout   = (float*)alloc((size_t)512 * 96 * 4);
  f16*   tT     = (f16*)  alloc((size_t)8 * 128 * 64 * 2);
  float* x1b    = (float*)alloc((size_t)8 * 96 * HWn * 4);
  f16*   Af16   = (f16*)  alloc((size_t)8 * 64 * HWn * 2);
  f16*   xatt   = (f16*)  alloc((size_t)M1n * 96 * 2 + 512);  // pad: B-tile overread
  float* logits = (float*)alloc((size_t)M1n * 64 * 4);
  f16*   catf   = (f16*)  alloc((size_t)M1n * 288 * 2);
  float* catc   = (float*)alloc((size_t)M1n * 96 * 4);
  float* resm   = (float*)alloc((size_t)M1n * 96 * 4);
  float* hmB    = (float*)alloc((size_t)M1n * 96 * 4);
  f16*   dw2h   = (f16*)  alloc((size_t)M1n * 96 * 2);
  float* pw2o   = (float*)alloc((size_t)M1n * 96 * 4);
  float* avg = (float*)alloc(8 * 96 * 4);
  float* a1b = (float*)alloc(8 * 12 * 4);
  float* cf  = (float*)alloc(8 * 96 * 4);
  float* s0  = (float*)alloc((size_t)8 * HWn * 4);
  float* s1b = (float*)alloc((size_t)8 * HWn * 4);
  float* s4a = (float*)alloc((size_t)8 * 4 * HWn * 4);
  float* s4b = (float*)alloc((size_t)8 * 4 * HWn * 4);
  float* pfB = (float*)alloc((size_t)8 * HWn * 4);
  f16* pw1W  = (f16*)alloc(96 * 128 * 2);
  f16* pw2W  = (f16*)alloc(96 * 128 * 2);
  f16* resMW = (f16*)alloc(96 * 128 * 2);
  f16* bnattW= (f16*)alloc(288 * 128 * 2);
  f16* assignW=(f16*)alloc(96 * 64 * 2);
  f16* inW   = (f16*)alloc(96 * 384 * 2);
  f16* xpW   = (f16*)alloc(192 * 64 * 2);
  f16* dtW   = (f16*)alloc(32 * 192 * 2);
  f16* outW  = (f16*)alloc(192 * 128 * 2);
  // mamba scratch pool (sized for rows = M1n, reused by all three passes)
  f16*   xln  = (f16*)  alloc((size_t)M1n * 96 * 2);
  float* xz   = (float*)alloc((size_t)M1n * 384 * 4);
  float* ub   = (float*)alloc((size_t)M1n * 192 * 4);
  f16*   uh   = (f16*)  alloc((size_t)M1n * 192 * 2);
  float* dbc  = (float*)alloc((size_t)M1n * 38 * 4);
  f16*   dtin = (f16*)  alloc((size_t)M1n * 32 * 2);
  float* Bmb  = (float*)alloc((size_t)M1n * 16 * 4);
  float* Cmb  = (float*)alloc((size_t)M1n * 16 * 4);
  float* dtb  = (float*)alloc((size_t)M1n * 192 * 4);
  f16*   yact = (f16*)  alloc((size_t)M1n * 192 * 2);

  auto nb = [](long long n) { return dim3((unsigned)((n + 255) / 256)); };

  auto pack = [&](const float* src, f16* dst, int R, int Cc, long long sr, long long sc,
                  int Rp, int Cp, long long sb, long long db, int batches) {
    long long tot = (long long)Rp * Cp;
    dim3 grid((unsigned)((tot + 255) / 256), 1, (unsigned)batches);
    k_pack<<<grid, dim3(256), 0, stream>>>(src, dst, R, Cc, sr, sc, Rp, Cp, sb, db);
  };

  auto gemm = [&](const f16* A, const f16* Bw, const float* bias, float* Co,
                  int M, int N, int K, int lda, int ldb, int ldc,
                  long long sA, long long sB, long long sC, int batches) {
    dim3 grid((unsigned)((M + 63) / 64), (unsigned)((N + 63) / 64), (unsigned)batches);
    k_gemm<<<grid, dim3(128), 0, stream>>>(A, Bw, bias, Co, M, N, K, lda, ldb, ldc, sA, sB, sC);
  };

  auto run_mamba = [&](const float* src, int rows, int rpb, int transposed,
                       const float* lng, const float* lnb, int pb, float* outRows) {
    int L = rpb;
    pack(P(pb + 0), inW,  96, 384, 1, 96,  96, 384, 0, 0, 1);   // in_proj^T
    pack(P(pb + 3), xpW,  192, 38, 1, 192, 192, 64, 0, 0, 1);   // x_proj^T (pad N 38->64)
    pack(P(pb + 4), dtW,  6, 192, 1, 6,   32, 192, 0, 0, 1);    // dt_proj^T (pad K 6->32)
    pack(P(pb + 8), outW, 192, 96, 1, 192, 192, 128, 0, 0, 1);  // out_proj^T
    k_layernorm<<<dim3((rows + 3) / 4), dim3(128), 0, stream>>>(src, xln, lng, lnb, rows, rpb, transposed);
    gemm(xln, inW, nullptr, xz, rows, 384, 96, 96, 384, 384, 0, 0, 0, 1);
    k_conv1d_silu<<<nb((long long)rows * 192), dim3(256), 0, stream>>>(xz, P(pb + 1), P(pb + 2), ub, uh, L);
    gemm(uh, xpW, nullptr, dbc, rows, 38, 192, 192, 64, 38, 0, 0, 0, 1);
    k_dtprep<<<nb((long long)rows * 64), dim3(256), 0, stream>>>(dbc, dtin, Bmb, Cmb, rows);
    gemm(dtin, dtW, P(pb + 5), dtb, rows, 192, 32, 32, 192, 192, 0, 0, 0, 1);
    k_softplus<<<nb((long long)rows * 192), dim3(256), 0, stream>>>(dtb, (long long)rows * 192);
    k_scan<<<dim3(192), dim3(128), 0, stream>>>(dtb, ub, Bmb, Cmb, xz + 192, P(pb + 6), P(pb + 7), yact, L);
    gemm(yact, outW, nullptr, outRows, rows, 96, 192, 192, 128, 96, 0, 0, 0, 1);
  };

  // ---- stem: dw1 -> pw1 -> bn1 -> h ----
  k_dw3x3_nchw<<<nb((long long)M1n * 96), dim3(256), 0, stream>>>(X, P(1), P(2), dw1h);
  pack(P(3), pw1W, 96, 96, 1, 96, 96, 128, 0, 0, 1);
  gemm(dw1h, pw1W, P(4), hrows, M1n, 96, 96, 96, 128, 96, 0, 0, 0, 1);
  k_bn_rows<<<nb((long long)M1n * 96), dim3(256), 0, stream>>>(hrows, P(5), P(6), (long long)M1n * 96, 96);
  pack(hrows, hf16, M1n, 96, 96, 1, M1n, 96, 0, 0, 1);

  // ---- two spatial mamba passes ----
  run_mamba(hrows, M1n, HWn, 0, P(13), P(14), 38, x0r);   // mamba0
  run_mamba(hrows, M1n, HWn, 1, P(15), P(16), 47, x01s);  // mamba01 (transposed scan)

  // ---- scanner attention ----
  k_chanmean<<<dim3(8 * 96), dim3(256), 0, stream>>>(hrows, avg);
  k_ca1<<<dim3(1), dim3(96), 0, stream>>>(avg, P(56), P(57), P(58), a1b);
  k_ca2<<<dim3(3), dim3(256), 0, stream>>>(a1b, P(59), P(60), cf);
  k_spatial_sum<<<nb(8 * HWn), dim3(256), 0, stream>>>(hrows, cf, s0);
  k_conv7<<<nb(8 * HWn), dim3(256), 0, stream>>>(s0, P(61), P(62), P(63), P(64), s1b);
  k_conv3_1to4<<<nb(8 * 4 * HWn), dim3(256), 0, stream>>>(s1b, P(65), s4a);
  k_conv3_4to4<<<nb(8 * 4 * HWn), dim3(256), 0, stream>>>(s4a, P(66), s4b, 2);
  k_conv3_4to4<<<nb(8 * 4 * HWn), dim3(256), 0, stream>>>(s4b, P(67), s4a, 4);
  k_conv1sig<<<nb(8 * HWn), dim3(256), 0, stream>>>(s4a, P(68), P(69), pfB);
  k_pack_xatt<<<nb((long long)M1n * 96), dim3(256), 0, stream>>>(hrows, cf, pfB, xatt);
  pack(P(70), assignW, 96, 64, 1, 96, 96, 64, 0, 0, 1);
  gemm(xatt, assignW, nullptr, logits, M1n, 64, 96, 96, 64, 64, 0, 0, 0, 1);
  k_softmax64<<<nb(M1n), dim3(256), 0, stream>>>(logits, Af16);
  // tokens[b] (64x96) = A_[b] (64x4096) * xatt[b] (4096x96)
  gemm(Af16, xatt, nullptr, tokens, 64, 96, 4096, 4096, 96, 96,
       (long long)64 * 4096, (long long)4096 * 96, (long long)64 * 96, 8);

  // ---- token mamba + scatter back ----
  run_mamba(tokens, 512, 64, 0, P(17), P(18), 29, tout);
  pack(tout, tT, 96, 64, 1, 96, 128, 64, (long long)64 * 96, (long long)128 * 64, 8);
  // x1[b] (96x4096) = t^T[b] (96x64) * A_[b] (64x4096)
  gemm(tT, Af16, nullptr, x1b, 96, 4096, 64, 64, 4096, 4096,
       (long long)128 * 64, (long long)64 * 4096, (long long)96 * 4096, 8);

  // ---- concat + bnatt conv + resM + tail ----
  k_build_cat<<<nb((long long)M1n * 288), dim3(256), 0, stream>>>(x0r, x01s, x1b, P(19), P(20), catf);
  pack(P(21), bnattW, 288, 96, 1, 288, 288, 128, 0, 0, 1);
  gemm(catf, bnattW, P(22), catc, M1n, 96, 288, 288, 128, 96, 0, 0, 0, 1);
  pack(P(25), resMW, 96, 96, 1, 96, 96, 128, 0, 0, 1);
  gemm(hf16, resMW, P(26), resm, M1n, 96, 96, 96, 128, 96, 0, 0, 0, 1);
  k_hm<<<nb((long long)M1n * 96), dim3(256), 0, stream>>>(resm, catc, P(27), P(28), P(23), P(24), hmB);
  k_dw3x3_rows<<<nb((long long)M1n * 96), dim3(256), 0, stream>>>(hmB, P(7), P(8), dw2h);
  pack(P(9), pw2W, 96, 96, 1, 96, 96, 128, 0, 0, 1);
  gemm(dw2h, pw2W, P(10), pw2o, M1n, 96, 96, 96, 128, 96, 0, 0, 0, 1);
  k_final<<<nb((long long)Bn * Cn * HWn), dim3(256), 0, stream>>>(pw2o, P(11), P(12), X, (float*)d_out);
}
